// CrossRaionAttention_18957985645082
// MI455X (gfx1250) — matrix operand order repaired and checked by reading, not compiled, not run on hardware
//
#include <hip/hip_runtime.h>

#define B_ 4
#define R_ 512
#define S_ 256
#define D_ 128
#define H_ 8
#define HD_ 16

typedef __attribute__((ext_vector_type(16))) __bf16 v16bf;
typedef __attribute__((ext_vector_type(8)))  float  v8f;

__device__ __forceinline__ __bf16 f2bf(float f) {
  unsigned u = __builtin_bit_cast(unsigned, f);
  unsigned short h = (unsigned short)(u >> 16);   // truncate; residual goes to lo
  return __builtin_bit_cast(__bf16, h);
}
__device__ __forceinline__ float bf2f(__bf16 b) {
  unsigned short h = __builtin_bit_cast(unsigned short, b);
  unsigned u = ((unsigned)h) << 16;
  return __builtin_bit_cast(float, u);
}
__device__ __forceinline__ void split2(float x, __bf16 &hi, __bf16 &lo) {
  hi = f2bf(x);
  lo = f2bf(x - bf2f(hi));
}
__device__ __forceinline__ v8f wmma_bf16(v16bf a, v16bf b, v8f c) {
  // (neg_a, A, neg_b, B, c_mod, C, reuse_a, reuse_b)
  return __builtin_amdgcn_wmma_f32_16x16x32_bf16(false, a, false, b, (short)0, c,
                                                 false, false);
}
__device__ __forceinline__ float gelu_exact(float x) {
  return 0.5f * x * (1.0f + erff(x * 0.70710678118654752f));
}

// ---------------------------------------------------------------------------
// Kernel 1: temporal pool. One block per (b,r). pooled = mean_S gelu(LN(xW+b)).
// x slab (256x128 f32) read exactly once. W staged in LDS as per-lane WMMA
// fragments (bf16 hi/lo -> fp32-accurate bf16x3 WMMA).
// Register-pressure-shaped: tile loops kept ROLLED (unroll(disable)) so only
// ONE acc tile is live; Y tiles parked in LDS between stats pass and
// normalize/GELU/mean pass. No VGPR spills, no vgpr-msb addressing.
// ---------------------------------------------------------------------------
__global__ __launch_bounds__(256) void pool_kernel(
    const float* __restrict__ x, const float* __restrict__ tp_w,
    const float* __restrict__ tp_b, const float* __restrict__ tp_g,
    const float* __restrict__ tp_beta, float* __restrict__ pooled) {
  __shared__ __bf16 wf_hi[16384];   // [kc(4)][t(8)][lane(32)][i(16)]
  __shared__ __bf16 wf_lo[16384];
  __shared__ float  sY[16384];      // [wave(8)][t(8)][lane(32)][j(8)] parked tiles
  __shared__ float  s_bias[128], s_g[128], s_beta[128], s_pool[128];

  const int tid = threadIdx.x;
  if (tid < 128) {
    s_bias[tid] = tp_b[tid];
    s_g[tid]    = tp_g[tid];
    s_beta[tid] = tp_beta[tid];
    s_pool[tid] = 0.0f;
  }
  for (int f = tid; f < 16384; f += 256) {
    int i  = f & 15;
    int ln = (f >> 4) & 31;
    int ct = f >> 9;            // kc*8 + t
    int t  = ct & 7;
    int kc = ct >> 3;
    int k  = kc * 32 + (ln >> 4) * 8 + (i >> 3) * 16 + (i & 7);   // B-frag: table order (8-runs, half*8)
    int n  = t * 16 + (ln & 15);
    __bf16 hi, lo; split2(tp_w[k * D_ + n], hi, lo);
    wf_hi[f] = hi; wf_lo[f] = lo;
  }
  __syncthreads();

  const int wave = tid >> 5;
  const int lane = tid & 31;
  const int hh   = lane >> 4;
  const int m    = lane & 15;
  const size_t slab = (size_t)blockIdx.x * (S_ * D_);
  float* myY = &sY[(wave * 8) * 256 + lane * 8];   // + t*256 per tile

#pragma clang loop unroll(disable)
  for (int sidx = 0; sidx < 2; ++sidx) {
    const int row0 = (wave + sidx * 8) * 16;       // 16-row strip of S
    // ---- A fragments for all 4 K-chunks (hi/lo split) ----
    v16bf a_hi[4], a_lo[4];
    const float* arow = x + slab + (size_t)(row0 + m) * D_;
#pragma unroll
    for (int kc = 0; kc < 4; ++kc) {
      const float* p0 = arow + kc * 32 + hh * 8;   // K run {0..7}+half*8
      const float* p1 = p0 + 16;                   // K run {16..23}+half*8
#pragma unroll
      for (int i = 0; i < 8; ++i) {
        __bf16 hi, lo;
        split2(p0[i], hi, lo); a_hi[kc][i]     = hi; a_lo[kc][i]     = lo;
        split2(p1[i], hi, lo); a_hi[kc][8 + i] = hi; a_lo[kc][8 + i] = lo;
      }
    }
    // ---- Phase 1: one tile at a time; bias + row partial stats; park in LDS ----
    float rs[8], rq[8];
#pragma unroll
    for (int j = 0; j < 8; ++j) { rs[j] = 0.f; rq[j] = 0.f; }
#pragma clang loop unroll(disable)
    for (int t = 0; t < 8; ++t) {               // rolled: caps reg pressure
      v8f c = {};
#pragma unroll
      for (int kc = 0; kc < 4; ++kc) {
        const int off = ((kc * 8 + t) * 32 + lane) * 16;
        v16bf bh = *(const v16bf*)&wf_hi[off];
        v16bf bl = *(const v16bf*)&wf_lo[off];
        c = wmma_bf16(a_lo[kc], bh, c);
        c = wmma_bf16(a_hi[kc], bl, c);
        c = wmma_bf16(a_hi[kc], bh, c);
      }
      const float bias = s_bias[t * 16 + m];
#pragma unroll
      for (int j = 0; j < 8; ++j) {
        float v = c[j] + bias;
        c[j] = v;
        rs[j] += v;
        rq[j] += v * v;
      }
      *(v8f*)(myY + t * 256) = c;               // park tile (same-wave LDS)
    }
    // ---- Row stats across the 16-lane half-group (row = j + 8*hh) ----
#pragma unroll
    for (int j = 0; j < 8; ++j) {
#pragma unroll
      for (int msk = 1; msk < 16; msk <<= 1) {
        rs[j] += __shfl_xor(rs[j], msk, 32);
        rq[j] += __shfl_xor(rq[j], msk, 32);
      }
    }
    float mu[8], rsig[8];
#pragma unroll
    for (int j = 0; j < 8; ++j) {
      mu[j] = rs[j] * (1.0f / 128.0f);
      float var = rq[j] * (1.0f / 128.0f) - mu[j] * mu[j];
      rsig[j] = rsqrtf(var + 1e-5f);
    }
    // ---- Phase 2: unpark, LN scale/shift + exact GELU + partial mean over S ----
#pragma clang loop unroll(disable)
    for (int t = 0; t < 8; ++t) {
      v8f c = *(const v8f*)(myY + t * 256);
      const float g  = s_g[t * 16 + m];
      const float bt = s_beta[t * 16 + m];
      float ps = 0.f;
#pragma unroll
      for (int j = 0; j < 8; ++j) {
        float v = (c[j] - mu[j]) * rsig[j] * g + bt;
        ps += gelu_exact(v);
      }
      ps += __shfl_xor(ps, 16, 32);   // combine halves (same output column)
      if (hh == 0) atomicAdd(&s_pool[t * 16 + m], ps);
    }
  }
  __syncthreads();
  if (tid < 128)
    pooled[(size_t)blockIdx.x * D_ + tid] = s_pool[tid] * (1.0f / (float)S_);
}

// ---------------------------------------------------------------------------
// Generic small GEMM: C[M,N] = act(A[M,K] @ W[K,N] + bias), bf16x3 WMMA.
// Block = 256 thr (8 waves), block owns 128 rows, wave owns a 16-row strip.
// Tile loop rolled -> one live acc tile; A fragments hoisted.
// ---------------------------------------------------------------------------
template <int K, int N, bool GELU>
__global__ __launch_bounds__(256) void gemm_kernel(
    const float* __restrict__ A, const float* __restrict__ W,
    const float* __restrict__ bias, float* __restrict__ C) {
  constexpr int KC = K / 32;
  constexpr int NT = N / 16;
  constexpr int WF = KC * NT * 512;
  __shared__ __bf16 wf_hi[WF];
  __shared__ __bf16 wf_lo[WF];
  __shared__ float s_bias[N];

  const int tid = threadIdx.x;
  for (int i = tid; i < N; i += 256) s_bias[i] = bias[i];
  for (int f = tid; f < WF; f += 256) {
    int i  = f & 15;
    int ln = (f >> 4) & 31;
    int ct = f >> 9;
    int t  = ct % NT;
    int kc = ct / NT;
    int k  = kc * 32 + (ln >> 4) * 8 + (i >> 3) * 16 + (i & 7);
    int n  = t * 16 + (ln & 15);
    __bf16 hi, lo; split2(W[k * N + n], hi, lo);
    wf_hi[f] = hi; wf_lo[f] = lo;
  }
  __syncthreads();

  const int wave = tid >> 5;
  const int lane = tid & 31;
  const int hh   = lane >> 4;
  const int m    = lane & 15;
  const int row0 = blockIdx.x * 128 + wave * 16;

  v16bf a_hi[KC], a_lo[KC];
  const float* arow = A + (size_t)(row0 + m) * K;
#pragma unroll
  for (int kc = 0; kc < KC; ++kc) {
    const float* p0 = arow + kc * 32 + hh * 8;
    const float* p1 = p0 + 16;
#pragma unroll
    for (int i = 0; i < 8; ++i) {
      __bf16 hi, lo;
      split2(p0[i], hi, lo); a_hi[kc][i]     = hi; a_lo[kc][i]     = lo;
      split2(p1[i], hi, lo); a_hi[kc][8 + i] = hi; a_lo[kc][8 + i] = lo;
    }
  }

#pragma clang loop unroll(disable)
  for (int t = 0; t < NT; ++t) {               // rolled: one live acc tile
    v8f c = {};
#pragma unroll
    for (int kc = 0; kc < KC; ++kc) {
      const int off = ((kc * NT + t) * 32 + lane) * 16;
      v16bf bh = *(const v16bf*)&wf_hi[off];
      v16bf bl = *(const v16bf*)&wf_lo[off];
      c = wmma_bf16(a_lo[kc], bh, c);
      c = wmma_bf16(a_hi[kc], bl, c);
      c = wmma_bf16(a_hi[kc], bh, c);
    }
    const int n0 = t * 16 + m;
    const float bn = s_bias[n0];
    float* crow = C + (size_t)(row0 + 8 * hh) * N + n0;
#pragma unroll
    for (int j = 0; j < 8; ++j) {
      float v = c[j] + bn;
      if (GELU) v = gelu_exact(v);
      crow[(size_t)j * N] = v;
    }
  }
}

// ---------------------------------------------------------------------------
// Kernel: attention per (b,h) with geo log-prior, online softmax. K/V in LDS.
// ---------------------------------------------------------------------------
__global__ __launch_bounds__(256) void attn_kernel(
    const float* __restrict__ q, const float* __restrict__ k,
    const float* __restrict__ v, const float* __restrict__ log_prior,
    const float* __restrict__ prior_scale, float* __restrict__ ctx) {
  __shared__ float ks[R_ * HD_];
  __shared__ float vs[R_ * HD_];
  const int b = blockIdx.x / H_;
  const int h = blockIdx.x % H_;
  const int tid = threadIdx.x;
  const float pscale = prior_scale[0];

  for (int i = tid; i < R_ * HD_; i += 256) {
    int r = i >> 4;
    int d = i & 15;
    size_t gi = ((size_t)(b * R_ + r)) * D_ + h * HD_ + d;
    ks[i] = k[gi];
    vs[i] = v[gi];
  }
  __syncthreads();

  for (int qi = tid; qi < R_; qi += 256) {
    float qv[16];
    const float* qp = q + ((size_t)(b * R_ + qi)) * D_ + h * HD_;
#pragma unroll
    for (int d = 0; d < 16; ++d) qv[d] = qp[d];
    const float* lp = log_prior + (size_t)qi * R_;
    float mx = -3.0e38f, l = 0.f;
    float acc[16];
#pragma unroll
    for (int d = 0; d < 16; ++d) acc[d] = 0.f;
    for (int key = 0; key < R_; ++key) {
      const float* kr = &ks[key * 16];
      float s = 0.f;
#pragma unroll
      for (int d = 0; d < 16; ++d) s += qv[d] * kr[d];
      s = s * 0.25f + pscale * lp[key];           // 1/sqrt(HD) = 0.25
      float nm   = fmaxf(mx, s);
      float corr = expf(mx - nm);
      float p    = expf(s - nm);
      l = l * corr + p;
      const float* vr = &vs[key * 16];
#pragma unroll
      for (int d = 0; d < 16; ++d) acc[d] = acc[d] * corr + p * vr[d];
      mx = nm;
    }
    const float inv_l = 1.0f / l;
    float* cp = ctx + ((size_t)(b * R_ + qi)) * D_ + h * HD_;
#pragma unroll
    for (int d = 0; d < 16; ++d) cp[d] = acc[d] * inv_l;
  }
}

// ---------------------------------------------------------------------------
// Kernel: out = LN(x + tb broadcast over S). One wave per 128-float row.
// ---------------------------------------------------------------------------
__global__ __launch_bounds__(256) void lnout_kernel(
    const float* __restrict__ x, const float* __restrict__ tb,
    const float* __restrict__ g, const float* __restrict__ be,
    float* __restrict__ out) {
  const int wave = threadIdx.x >> 5;
  const int lane = threadIdx.x & 31;
  const size_t row = (size_t)blockIdx.x * 8 + wave;   // < B*R*S
  const size_t br  = row / S_;
  float4 xv = ((const float4*)(x + row * D_))[lane];
  float4 tv = ((const float4*)(tb + br * D_))[lane];
  float v0 = xv.x + tv.x, v1 = xv.y + tv.y, v2 = xv.z + tv.z, v3 = xv.w + tv.w;
  float s  = v0 + v1 + v2 + v3;
  float sq = v0 * v0 + v1 * v1 + v2 * v2 + v3 * v3;
#pragma unroll
  for (int m = 1; m < 32; m <<= 1) {
    s  += __shfl_xor(s,  m, 32);
    sq += __shfl_xor(sq, m, 32);
  }
  float mu   = s * (1.0f / 128.0f);
  float var  = sq * (1.0f / 128.0f) - mu * mu;
  float rsig = rsqrtf(var + 1e-5f);
  float4 gv = ((const float4*)g)[lane];
  float4 bv = ((const float4*)be)[lane];
  float4 o;
  o.x = (v0 - mu) * rsig * gv.x + bv.x;
  o.y = (v1 - mu) * rsig * gv.y + bv.y;
  o.z = (v2 - mu) * rsig * gv.z + bv.z;
  o.w = (v3 - mu) * rsig * gv.w + bv.w;
  ((float4*)(out + row * D_))[lane] = o;
}

// ---------------------------------------------------------------------------
extern "C" void kernel_launch(void* const* d_in, const int* in_sizes, int n_in,
                              void* d_out, int out_size, void* d_ws, size_t ws_size,
                              hipStream_t stream) {
  (void)in_sizes; (void)n_in; (void)out_size; (void)ws_size;
  const float* x      = (const float*)d_in[0];
  const float* lp     = (const float*)d_in[1];
  const float* tpw    = (const float*)d_in[2];
  const float* tpb    = (const float*)d_in[3];
  const float* tpg    = (const float*)d_in[4];
  const float* tpbeta = (const float*)d_in[5];
  const float* wq     = (const float*)d_in[6];
  const float* bq     = (const float*)d_in[7];
  const float* wk     = (const float*)d_in[8];
  const float* bk     = (const float*)d_in[9];
  const float* wv     = (const float*)d_in[10];
  const float* bv     = (const float*)d_in[11];
  const float* wo     = (const float*)d_in[12];
  const float* bo     = (const float*)d_in[13];
  const float* pscale = (const float*)d_in[14];
  const float* lng    = (const float*)d_in[15];
  const float* lnb    = (const float*)d_in[16];
  const float* w1     = (const float*)d_in[17];
  const float* b1     = (const float*)d_in[18];
  const float* w2     = (const float*)d_in[19];
  const float* b2     = (const float*)d_in[20];
  float* out = (float*)d_out;

  const int BR = B_ * R_;                 // 2048
  float* ws     = (float*)d_ws;           // 9 MB of f32 scratch used
  float* pooled = ws;                     // BR*128
  float* qb     = pooled + BR * 128;
  float* kb     = qb + BR * 128;
  float* vb     = kb + BR * 128;
  float* ctx    = vb + BR * 128;
  float* cross  = ctx + BR * 128;
  float* hbuf   = cross + BR * 128;       // BR*256
  float* tbuf   = hbuf + BR * 256;        // BR*128

  pool_kernel<<<BR, 256, 0, stream>>>(x, tpw, tpb, tpg, tpbeta, pooled);
  gemm_kernel<128, 128, false><<<BR / 128, 256, 0, stream>>>(pooled, wq, bq, qb);
  gemm_kernel<128, 128, false><<<BR / 128, 256, 0, stream>>>(pooled, wk, bk, kb);
  gemm_kernel<128, 128, false><<<BR / 128, 256, 0, stream>>>(pooled, wv, bv, vb);
  attn_kernel<<<B_ * H_, 256, 0, stream>>>(qb, kb, vb, lp, pscale, ctx);
  gemm_kernel<128, 128, false><<<BR / 128, 256, 0, stream>>>(ctx, wo, bo, cross);
  gemm_kernel<128, 256, true ><<<BR / 128, 256, 0, stream>>>(cross, w1, b1, hbuf);
  gemm_kernel<256, 128, false><<<BR / 128, 256, 0, stream>>>(hbuf, w2, b2, tbuf);
  lnout_kernel<<<(B_ * R_ * S_) / 8, 256, 0, stream>>>(x, tbuf, lng, lnb, out);
}
